// Encoder_7679401525515
// MI455X (gfx1250) — matrix operand order repaired and checked by reading, not compiled, not run on hardware
//
#include <hip/hip_runtime.h>

#define BB   64
#define TT   1024
#define HH   512
#define GG   2048
#define FF   128
#define RING 8

typedef __attribute__((ext_vector_type(16))) __bf16 v16bf;
typedef __attribute__((ext_vector_type(8)))  float  v8f;

struct Q32 { int4 a, b; };
static_assert(sizeof(Q32) == 32, "Q32 size");
static_assert(sizeof(v16bf) == 32, "v16bf size");

// ---------------- workspace layout (bytes) ----------------
constexpr size_t SZ_W0    = 128ull * 20 * 1024;          // layer0 B-tiles: nb(128) x kb(20) x 1KB
constexpr size_t SZ_W1    = 128ull * 32 * 1024;          // layer1 B-tiles: nb(128) x kb(32) x 1KB
constexpr size_t SZ_XB    = (size_t)TT * 16 * 1024;      // x as A-tiles: t x (m4 x kb4) x 1KB
constexpr size_t SZ_RING  = (size_t)RING * 64 * 1024;    // h0 A-tiles: slot x (m4 x kb16) x 1KB
constexpr size_t OFF_W0   = 0;
constexpr size_t OFF_W1   = OFF_W0 + SZ_W0;
constexpr size_t OFF_XB   = OFF_W1 + SZ_W1;
constexpr size_t OFF_RING = OFF_XB + SZ_XB;
constexpr size_t OFF_B0   = OFF_RING + SZ_RING;
constexpr size_t OFF_B1   = OFF_B0 + GG * 4;
constexpr size_t OFF_FLG  = OFF_B1 + GG * 4;             // [0]=prod, [16]=cons (64B apart)

// ---------------- helpers ----------------
__device__ __host__ __forceinline__ unsigned short f2bf(float x) {
    unsigned u = __builtin_bit_cast(unsigned, x);
    unsigned r = u + 0x7FFFu + ((u >> 16) & 1u);   // round-to-nearest-even
    return (unsigned short)(r >> 16);
}

// XOR-swizzled LDS index for the 64x512 bf16 h buffer: 16B chunks permuted per
// row so ds_load_b128 across 16 rows hits 64 distinct banks.
__device__ __forceinline__ int lds_idx(int r, int c) {
    return r * 512 + ((((c >> 3) ^ (r & 63)) << 3) | (c & 7));
}

__device__ __forceinline__ v16bf load2(const unsigned short* p0, const unsigned short* p1) {
    Q32 q;
    q.a = *(const int4*)p0;
    q.b = *(const int4*)p1;
    return __builtin_bit_cast(v16bf, q);
}

__device__ __forceinline__ float fsig(float x)  { return 1.f / (1.f + __expf(-x)); }
__device__ __forceinline__ float ftanh(float x) { return 2.f * fsig(2.f * x) - 1.f; }

// ---------------- prep kernels ----------------
__global__ void prep_misc(const float* bih0, const float* bhh0,
                          const float* bih1, const float* bhh1, unsigned char* ws) {
    int i = blockIdx.x * blockDim.x + threadIdx.x;
    float* b0 = (float*)(ws + OFF_B0);
    float* b1 = (float*)(ws + OFF_B1);
    if (i < GG) { b0[i] = bih0[i] + bhh0[i]; b1[i] = bih1[i] + bhh1[i]; }
    if (i < 32) ((unsigned int*)(ws + OFF_FLG))[i] = 0u;
}

// Pack [Whh^T ; Wih^T] (K x N) into WMMA B-fragment tiles, bf16.
// Tile (nb, kb): lane l holds N = nb*16 + (l%16), K = kb*32 + (l<16?0:8) + e + (e>=8?8:0), e=0..15.
__global__ void prep_packW(const float* Whh0, const float* Wih0,
                           const float* Whh1, const float* Wih1, unsigned char* ws) {
    long idx = (long)blockIdx.x * blockDim.x + threadIdx.x;
    const long n0 = 128L * 20 * 512;
    const long n1 = 128L * 32 * 512;
    if (idx >= n0 + n1) return;
    int layer1 = idx >= n0;
    long r = layer1 ? idx - n0 : idx;
    unsigned char* dst = ws + (layer1 ? OFF_W1 : OFF_W0);
    int KB = layer1 ? 32 : 20;
    const float* Wh = layer1 ? Whh1 : Whh0;
    const float* Wi = layer1 ? Wih1 : Wih0;
    int  e    = (int)(r & 15);
    int  lane = (int)((r >> 4) & 31);
    long tile = r >> 9;
    int  kb   = (int)(tile % KB);
    int  nb   = (int)(tile / KB);
    int  K    = kb * 32 + (lane < 16 ? 0 : 8) + e + (e >= 8 ? 8 : 0);
    int  N    = nb * 16 + (lane & 15);
    float v;
    if (K < 512) v = Wh[(long)N * 512 + K];                                   // recurrent part
    else         v = layer1 ? Wi[(long)N * 512 + (K - 512)]                   // h0 input (K=512)
                            : Wi[(long)N * 128 + (K - 512)];                  // x input (F=128)
    ((unsigned short*)dst)[tile * 512 + lane * 16 + e] = f2bf(v);
}

// Pack x (B,F,T) into per-timestep WMMA A-fragment tiles, bf16.
// Tile (t, m, kb): lane l holds M-row = m*16 + (l%16);
// K = kb*32 + Krel, Krel = e + (e>=8?8:0) + (l>=16?8:0).
__global__ void prep_packX(const float* x, unsigned char* ws) {
    long idx = (long)blockIdx.x * blockDim.x + threadIdx.x;
    if (idx >= (long)TT * 8192) return;
    int e    = (int)(idx & 15);
    int lane = (int)((idx >> 4) & 31);
    int kb   = (int)((idx >> 9) & 3);
    int m    = (int)((idx >> 11) & 3);
    int t    = (int)(idx >> 13);
    int Krel = e + (e >= 8 ? 8 : 0) + (lane >= 16 ? 8 : 0);
    int f    = kb * 32 + Krel;
    int row  = m * 16 + (lane & 15);
    float v  = x[((long)row * FF + f) * TT + t];
    unsigned short* dst = (unsigned short*)(ws + OFF_XB);
    dst[(((long)t * 4 + m) * 4 + kb) * 512 + lane * 16 + e] = f2bf(v);
}

// ---------------- persistent per-layer LSTM block ----------------
template <int LAYER>
__device__ void run_layer(unsigned char* ws, float* out, unsigned short* hbuf) {
    const int tid  = threadIdx.x;
    const int lane = tid & 31;
    const int w    = tid >> 5;        // wave 0..15, owns h columns [w*32, w*32+32)
    const int lrow = lane & 15;
    const int hi   = lane >> 4;       // lane half
    const int aoff = hi * 8;          // A-frag K sub-offset
    const int KB   = (LAYER == 0) ? 20 : 32;

    const unsigned char* wpack = ws + (LAYER == 0 ? OFF_W0 : OFF_W1);
    const float*         bvec  = (const float*)(ws + (LAYER == 0 ? OFF_B0 : OFF_B1));
    const unsigned char* xb    = ws + OFF_XB;
    unsigned char*       ring  = ws + OFF_RING;
    unsigned int*        prod  = (unsigned int*)(ws + OFF_FLG);
    unsigned int*        cons  = (unsigned int*)(ws + OFF_FLG) + 16;

    for (int i = tid; i < BB * HH; i += 512) hbuf[i] = 0;   // h = 0
    __syncthreads();

    const int col = w * 32;
    float bias[2][4];
#pragma unroll
    for (int cb = 0; cb < 2; ++cb)
#pragma unroll
        for (int q = 0; q < 4; ++q)
            bias[cb][q] = bvec[q * 512 + col + cb * 16 + lrow];

    v8f c_st[4][2];
#pragma unroll
    for (int m = 0; m < 4; ++m)
#pragma unroll
        for (int cb = 0; cb < 2; ++cb)
            c_st[m][cb] = (v8f){0, 0, 0, 0, 0, 0, 0, 0};

    unsigned int hpack[4][2][4];   // staged new h (packed bf16 pairs)

    for (int t = 0; t < TT; ++t) {
        if (LAYER == 1) {
            if (tid == 0)
                while (__hip_atomic_load(prod, __ATOMIC_ACQUIRE, __HIP_MEMORY_SCOPE_AGENT) < (unsigned)(t + 1))
                    __builtin_amdgcn_s_sleep(2);
            __syncthreads();
        } else if (t >= RING) {
            if (tid == 0)
                while (__hip_atomic_load(cons, __ATOMIC_ACQUIRE, __HIP_MEMORY_SCOPE_AGENT) < (unsigned)(t - RING + 1))
                    __builtin_amdgcn_s_sleep(2);
            __syncthreads();
        }

        const unsigned char* astep = (LAYER == 0)
            ? xb   + ((size_t)t * 16) * 1024
            : ring + ((size_t)(t & (RING - 1)) * 64) * 1024;

#pragma unroll 1
        for (int m = 0; m < 4; ++m) {
            const int r = m * 16 + lrow;
#pragma unroll 1
            for (int cb = 0; cb < 2; ++cb) {
                const int nbb = w * 2 + cb;
                const unsigned char* bbase = wpack + ((size_t)nbb * KB) * 1024 + lane * 32;

                auto loadB = [&](int q, int kb) -> v16bf {
                    const unsigned char* bp = bbase + ((size_t)(q * 32 * KB + kb)) * 1024;
                    return load2((const unsigned short*)bp, (const unsigned short*)(bp + 16));
                };
                auto loadA_lds = [&](int kb) -> v16bf {
                    int c0 = kb * 32 + aoff;
                    return load2(&hbuf[lds_idx(r, c0)], &hbuf[lds_idx(r, c0 + 16)]);
                };
                auto loadA_g = [&](int kb) -> v16bf {
                    const unsigned char* p =
                        astep + ((size_t)(m * 16 + (kb - 16))) * 1024 + lane * 32;
                    return load2((const unsigned short*)p, (const unsigned short*)(p + 16));
                };

                v8f acc[4];
#pragma unroll
                for (int q = 0; q < 4; ++q) acc[q] = (v8f){0, 0, 0, 0, 0, 0, 0, 0};

                // prime the pipeline: first B set + first global-phase A
                v16bf bcur[4], bnxt[4];
#pragma unroll
                for (int q = 0; q < 4; ++q) bcur[q] = loadB(q, 0);
                v16bf ag = loadA_g(16);

                // ---- phase 1: kb = 0..15, A from LDS, B double-buffered ----
#pragma unroll 2
                for (int kb = 0; kb < 16; ++kb) {
                    v16bf a = loadA_lds(kb);
#pragma unroll
                    for (int q = 0; q < 4; ++q) bnxt[q] = loadB(q, kb + 1);
#pragma unroll
                    for (int q = 0; q < 4; ++q)
                        acc[q] = __builtin_amdgcn_wmma_f32_16x16x32_bf16(
                            false, a, false, bcur[q], (short)0, acc[q], false, false);
#pragma unroll
                    for (int q = 0; q < 4; ++q) bcur[q] = bnxt[q];
                }

                // ---- phase 2: kb = 16..KB-2, A from global, both pipelined ----
#pragma unroll 2
                for (int kb = 16; kb < KB - 1; ++kb) {
                    v16bf a = ag;
                    ag = loadA_g(kb + 1);
#pragma unroll
                    for (int q = 0; q < 4; ++q) bnxt[q] = loadB(q, kb + 1);
#pragma unroll
                    for (int q = 0; q < 4; ++q)
                        acc[q] = __builtin_amdgcn_wmma_f32_16x16x32_bf16(
                            false, a, false, bcur[q], (short)0, acc[q], false, false);
#pragma unroll
                    for (int q = 0; q < 4; ++q) bcur[q] = bnxt[q];
                }

                // ---- epilogue: kb = KB-1, no further prefetch ----
#pragma unroll
                for (int q = 0; q < 4; ++q)
                    acc[q] = __builtin_amdgcn_wmma_f32_16x16x32_bf16(
                        false, ag, false, bcur[q], (short)0, acc[q], false, false);

                // gate nonlinearity + cell update (f32)
#pragma unroll
                for (int j = 0; j < 8; ++j) {
                    float gi = fsig (acc[0][j] + bias[cb][0]);
                    float gf = fsig (acc[1][j] + bias[cb][1]);
                    float gg = ftanh(acc[2][j] + bias[cb][2]);
                    float go = fsig (acc[3][j] + bias[cb][3]);
                    float cc = gf * c_st[m][cb][j] + gi * gg;
                    c_st[m][cb][j] = cc;
                    float hh = go * ftanh(cc);
                    unsigned int hb = f2bf(hh);
                    if (j & 1) hpack[m][cb][j >> 1] |= hb << 16;
                    else       hpack[m][cb][j >> 1]  = hb;
                    if (LAYER == 1 && t == TT - 1)
                        out[(size_t)(m * 16 + j + hi * 8) * HH + col + cb * 16 + lrow] = hh;
                }
            }
        }
        __syncthreads();             // all reads of old h complete

        // commit new h to LDS (D-tile layout: row = m*16 + j + hi*8, col = lane col)
#pragma unroll
        for (int m = 0; m < 4; ++m)
#pragma unroll
            for (int cb = 0; cb < 2; ++cb)
#pragma unroll
                for (int j = 0; j < 8; ++j) {
                    int rr = m * 16 + j + hi * 8;
                    int cc = col + cb * 16 + lrow;
                    unsigned int pk = hpack[m][cb][j >> 1];
                    hbuf[lds_idx(rr, cc)] =
                        (j & 1) ? (unsigned short)(pk >> 16) : (unsigned short)pk;
                }
        __syncthreads();             // new h visible to whole workgroup

        if (LAYER == 0) {
            // publish h0_t as A-fragment tiles (m, kb=w) into the ring
            unsigned char* slot = ring + ((size_t)(t & (RING - 1)) * 64) * 1024;
#pragma unroll
            for (int m = 0; m < 4; ++m) {
                int r2 = m * 16 + lrow;
                int c0 = w * 32 + aoff;
                int4 qa = *(const int4*)&hbuf[lds_idx(r2, c0)];
                int4 qb = *(const int4*)&hbuf[lds_idx(r2, c0 + 16)];
                unsigned char* dp = slot + ((size_t)(m * 16 + w)) * 1024 + lane * 32;
                *(int4*)dp        = qa;
                *(int4*)(dp + 16) = qb;
            }
            __threadfence();
            __syncthreads();
            if (tid == 0)
                __hip_atomic_store(prod, (unsigned)(t + 1), __ATOMIC_RELEASE, __HIP_MEMORY_SCOPE_AGENT);
        } else {
            if (tid == 0)
                __hip_atomic_store(cons, (unsigned)(t + 1), __ATOMIC_RELEASE, __HIP_MEMORY_SCOPE_AGENT);
        }
    }
}

__global__ __launch_bounds__(512, 1) void lstm_main(unsigned char* ws, float* out) {
    __shared__ unsigned short hbuf[BB * HH];   // 64KB: h state, bf16, XOR-swizzled
    if (blockIdx.x == 0) run_layer<0>(ws, out, hbuf);
    else                 run_layer<1>(ws, out, hbuf);
}

// ---------------- host launch ----------------
extern "C" void kernel_launch(void* const* d_in, const int* in_sizes, int n_in,
                              void* d_out, int out_size, void* d_ws, size_t ws_size,
                              hipStream_t stream) {
    const float* x    = (const float*)d_in[0];
    const float* Wih0 = (const float*)d_in[1];
    const float* Whh0 = (const float*)d_in[2];
    const float* bih0 = (const float*)d_in[3];
    const float* bhh0 = (const float*)d_in[4];
    const float* Wih1 = (const float*)d_in[5];
    const float* Whh1 = (const float*)d_in[6];
    const float* bih1 = (const float*)d_in[7];
    const float* bhh1 = (const float*)d_in[8];
    unsigned char* ws  = (unsigned char*)d_ws;
    float*         out = (float*)d_out;

    prep_misc<<<(GG + 255) / 256, 256, 0, stream>>>(bih0, bhh0, bih1, bhh1, ws);

    long nW = (128L * 20 + 128L * 32) * 512;
    prep_packW<<<(int)((nW + 255) / 256), 256, 0, stream>>>(Whh0, Wih0, Whh1, Wih1, ws);

    long nX = (long)TT * 8192;
    prep_packX<<<(int)((nX + 255) / 256), 256, 0, stream>>>(x, ws);

    lstm_main<<<2, 512, 0, stream>>>(ws, out);
}
